// DifferentialGPT_72490458022174
// MI455X (gfx1250) — matrix operand order repaired and checked by reading, not compiled, not run on hardware
//
#include <hip/hip_runtime.h>
#include <stdint.h>

// ---------------- problem dimensions ----------------
constexpr int Vv  = 50257;
constexpr int Cc  = 1024;
constexpr int Hh  = 16;
constexpr int HSs = 64;
constexpr int Dd  = 32;
constexpr int Ll  = 4;
constexpr int Bb  = 2;
constexpr int Tt  = 1024;
constexpr int MT  = Bb * Tt;   // 2048 tokens
constexpr int C2  = 2 * Cc;    // 2048
constexpr int C3  = 3 * Cc;    // 3072 (fused qkv)
constexpr int C4  = 4 * Cc;    // 4096
constexpr float EPSf = 1e-5f;

// ---------------- WMMA types ----------------
typedef __attribute__((ext_vector_type(16))) __bf16 v16bf;
typedef __attribute__((ext_vector_type(8)))  float  v8f;

union Frag { v16bf v; uint4 q[2]; };

static __device__ __forceinline__ v8f wmma_bf16(v16bf a, v16bf b, v8f c) {
  return __builtin_amdgcn_wmma_f32_16x16x32_bf16(false, a, false, b, (short)0, c, false, false);
}

static __device__ __forceinline__ unsigned short f32_to_bf16(float f) {
  union { float f; unsigned u; } v; v.f = f;
  unsigned u = v.u;
  u = u + 0x7fffu + ((u >> 16) & 1u);   // round-to-nearest-even
  return (unsigned short)(u >> 16);
}

// ---------------- CDNA5 async global->LDS copy (ASYNCcnt path) ----------------
#if defined(__HIP_DEVICE_COMPILE__)
#if __has_builtin(__builtin_amdgcn_global_load_async_to_lds_b128)
#define ASYNC_LDS 1
#endif
#endif

typedef int v4i_gcc __attribute__((vector_size(16)));
typedef __attribute__((address_space(1))) v4i_gcc* as1_v4i;
typedef __attribute__((address_space(3))) v4i_gcc* as3_v4i;
typedef __attribute__((address_space(1))) void* as1_void;
typedef __attribute__((address_space(3))) void* as3_void;

static __device__ __forceinline__ void async_copy16(const void* g, void* l) {
#ifdef ASYNC_LDS
  __builtin_amdgcn_global_load_async_to_lds_b128(
      (as1_v4i)(as1_void)(void*)g, (as3_v4i)(as3_void)l, 0, 0);
#else
  *(uint4*)l = *(const uint4*)g;
#endif
}

static __device__ __forceinline__ void wait_async() {
#ifdef ASYNC_LDS
#if __has_builtin(__builtin_amdgcn_s_wait_asynccnt)
  __builtin_amdgcn_s_wait_asynccnt(0);
#else
  asm volatile("s_wait_asynccnt 0x0" ::: "memory");
#endif
#endif
}

// ---------------- elementwise / conversion kernels ----------------

__global__ void cvt_bf16_kernel(const float* __restrict__ in,
                                unsigned short* __restrict__ out, long long n) {
  long long i = (long long)blockIdx.x * blockDim.x + threadIdx.x;
  long long stride = (long long)gridDim.x * blockDim.x;
  for (; i < n; i += stride) out[i] = f32_to_bf16(in[i]);
}

// in: f32 [K,N] row-major  ->  out: bf16 [N,K] row-major.  K,N multiples of 32.
__global__ __launch_bounds__(256)
void transpose_cvt_kernel(const float* __restrict__ in,
                          unsigned short* __restrict__ out, int K, int N) {
  __shared__ float tile[32][33];
  const int n0 = blockIdx.x * 32, k0 = blockIdx.y * 32;
  const int tx = threadIdx.x & 31, ty = threadIdx.x >> 5;   // 32 x 8
#pragma unroll
  for (int r = 0; r < 4; ++r) {
    int k = k0 + ty + r * 8;
    tile[ty + r * 8][tx] = in[(size_t)k * N + n0 + tx];
  }
  __syncthreads();
#pragma unroll
  for (int r = 0; r < 4; ++r) {
    int n = n0 + ty + r * 8;
    out[(size_t)n * K + k0 + tx] = f32_to_bf16(tile[tx][ty + r * 8]);
  }
}

__global__ __launch_bounds__(256)
void embed_kernel(const int* __restrict__ idx, const float* __restrict__ wte,
                  const float* __restrict__ wpe, float* __restrict__ x) {
  const int token = blockIdx.x;
  const int id = idx[token];
  float4 a = ((const float4*)(wte + (size_t)id * Cc))[threadIdx.x];
  float4 p = ((const float4*)(wpe + (size_t)id * Cc))[threadIdx.x];
  float4 o = {a.x + p.x, a.y + p.y, a.z + p.z, a.w + p.w};
  ((float4*)(x + (size_t)token * Cc))[threadIdx.x] = o;
}

__global__ __launch_bounds__(256)
void rmsnorm_kernel(const float* __restrict__ x, const float* __restrict__ w,
                    unsigned short* __restrict__ out) {
  __shared__ float red[256];
  const int row = blockIdx.x, t = threadIdx.x;
  float4 v = ((const float4*)(x + (size_t)row * Cc))[t];
  red[t] = v.x * v.x + v.y * v.y + v.z * v.z + v.w * v.w;
  __syncthreads();
  for (int s = 128; s > 0; s >>= 1) {
    if (t < s) red[t] += red[t + s];
    __syncthreads();
  }
  const float r = rsqrtf(red[0] / (float)Cc + EPSf);
  float4 ww = ((const float4*)w)[t];
  size_t o = (size_t)row * Cc + t * 4;
  out[o + 0] = f32_to_bf16(v.x * r * ww.x);
  out[o + 1] = f32_to_bf16(v.y * r * ww.y);
  out[o + 2] = f32_to_bf16(v.z * r * ww.z);
  out[o + 3] = f32_to_bf16(v.w * r * ww.w);
}

__global__ __launch_bounds__(256)
void final_ln_kernel(const float* __restrict__ x, const float* __restrict__ w,
                     unsigned short* __restrict__ out) {
  __shared__ float red[256];
  const int row = blockIdx.x, t = threadIdx.x;
  float4 v = ((const float4*)(x + (size_t)row * Cc))[t];
  red[t] = v.x + v.y + v.z + v.w;
  __syncthreads();
  for (int s = 128; s > 0; s >>= 1) {
    if (t < s) red[t] += red[t + s];
    __syncthreads();
  }
  const float mu = red[0] / (float)Cc;
  __syncthreads();
  float4 c = {v.x - mu, v.y - mu, v.z - mu, v.w - mu};
  red[t] = c.x * c.x + c.y * c.y + c.z * c.z + c.w * c.w;
  __syncthreads();
  for (int s = 128; s > 0; s >>= 1) {
    if (t < s) red[t] += red[t + s];
    __syncthreads();
  }
  const float r = rsqrtf(red[0] / (float)Cc + EPSf);
  float4 ww = ((const float4*)w)[t];
  size_t o = (size_t)row * Cc + t * 4;
  out[o + 0] = f32_to_bf16(c.x * r * ww.x);
  out[o + 1] = f32_to_bf16(c.y * r * ww.y);
  out[o + 2] = f32_to_bf16(c.z * r * ww.z);
  out[o + 3] = f32_to_bf16(c.w * r * ww.w);
}

__global__ void lam_kernel(const float* __restrict__ lq1, const float* __restrict__ lq2,
                           const float* __restrict__ lk1, const float* __restrict__ lk2,
                           float* __restrict__ lam) {
  const int i = threadIdx.x;
  if (i >= Ll * Hh) return;
  const int l = i / Hh;
  float d1 = 0.f, d2 = 0.f;
#pragma unroll
  for (int d = 0; d < Dd; ++d) {
    d1 += lq1[i * Dd + d] * lk1[i * Dd + d];
    d2 += lq2[i * Dd + d] * lk2[i * Dd + d];
  }
  const float li = 0.8f - 0.6f * __expf(-0.3f * ((float)l - 1.0f));
  lam[i] = __expf(d1) - __expf(d2) + li;
}

__global__ void swiglu_kernel(const float* __restrict__ u, unsigned short* __restrict__ p) {
  size_t i = (size_t)blockIdx.x * blockDim.x + threadIdx.x;
  if (i >= (size_t)MT * C2) return;
  size_t m = i / C2, j = i % C2;
  float g = u[m * C4 + j];
  float a = u[m * C4 + C2 + j];
  float s = a / (1.0f + __expf(-a));
  p[i] = f32_to_bf16(s * g);
}

// ---------------- tiled bf16 WMMA GEMM (double-buffered async LDS) ----------------
// D[M,N] = A[M,K](bf16,row-major) x Bt[N,K](bf16,row-major)^T  (+bias) (+=out)
template <bool ADD, bool BIAS, bool OUTBF>
__global__ __launch_bounds__(256)
void gemm_kernel(const unsigned short* __restrict__ A,
                 const unsigned short* __restrict__ Bt,
                 const float* __restrict__ bias,
                 float* __restrict__ outF,
                 unsigned short* __restrict__ outB,
                 int M, int N, int K) {
  __shared__ __align__(16) unsigned short As[2][128 * 64];   // [buf][row][k] K-step 64
  __shared__ __align__(16) unsigned short Bs[2][128 * 64];
  const int t = threadIdx.x;
  const int lane = t & 31, w = t >> 5;
  const int wm = w >> 1, wn = w & 1;                 // 4 x 2 wave grid
  const int lh = lane >> 4, ll = lane & 15;
  const int mBase = blockIdx.y * 128;
  const int nBase = blockIdx.x * 128;

  v8f acc[2][4];
#pragma unroll
  for (int mi = 0; mi < 2; ++mi)
#pragma unroll
    for (int ni = 0; ni < 4; ++ni) acc[mi][ni] = v8f{};

  // stage a 128x64 bf16 tile of A and of Bt into LDS buffer bi (1024 x 16B chunks each)
  auto stage = [&](int k0, int bi) {
#pragma unroll
    for (int j = 0; j < 4; ++j) {
      const int c = t * 4 + j;                       // chunk id
      const int row = c >> 3, ck = (c & 7) * 8;
      async_copy16(&A[(size_t)(mBase + row) * K + k0 + ck], &As[bi][row * 64 + ck]);
      int nrow = nBase + row; if (nrow >= N) nrow = N - 1;
      async_copy16(&Bt[(size_t)nrow * K + k0 + ck], &Bs[bi][row * 64 + ck]);
    }
  };

  stage(0, 0);
  wait_async();
  __syncthreads();

  const int nk = K >> 6;
  for (int kk = 0; kk < nk; ++kk) {
    const int bi = kk & 1;
    if (kk + 1 < nk) stage((kk + 1) << 6, bi ^ 1);   // prefetch while computing
    const unsigned short* as = &As[bi][0];
    const unsigned short* bs = &Bs[bi][0];
#pragma unroll
    for (int s = 0; s < 2; ++s) {                    // two 32-wide K slices
      Frag a[2], b[4];
#pragma unroll
      for (int mi = 0; mi < 2; ++mi) {
        const int r = wm * 32 + mi * 16 + ll;
        a[mi].q[0] = *(const uint4*)(&as[r * 64 + s * 32 + lh * 8]);
        a[mi].q[1] = *(const uint4*)(&as[r * 64 + s * 32 + lh * 8 + 16]);
      }
#pragma unroll
      for (int ni = 0; ni < 4; ++ni) {
        const int r = wn * 64 + ni * 16 + ll;
        b[ni].q[0] = *(const uint4*)(&bs[r * 64 + s * 32 + lh * 8]);
        b[ni].q[1] = *(const uint4*)(&bs[r * 64 + s * 32 + lh * 8 + 16]);
      }
#pragma unroll
      for (int mi = 0; mi < 2; ++mi)
#pragma unroll
        for (int ni = 0; ni < 4; ++ni)
          acc[mi][ni] = wmma_bf16(a[mi].v, b[ni].v, acc[mi][ni]);
    }
    wait_async();
    __syncthreads();
  }

#pragma unroll
  for (int mi = 0; mi < 2; ++mi)
#pragma unroll
    for (int ni = 0; ni < 4; ++ni) {
      const int n = nBase + wn * 64 + ni * 16 + ll;
      float bv = 0.f;
      if (BIAS) bv = bias[n < N ? n : (N - 1)];
      if (n < N) {
#pragma unroll
        for (int i = 0; i < 8; ++i) {
          const int m = mBase + wm * 32 + mi * 16 + i + 8 * lh;
          const float val = acc[mi][ni][i] + bv;
          if (OUTBF) {
            outB[(size_t)m * N + n] = f32_to_bf16(val);
          } else if (ADD) {
            outF[(size_t)m * N + n] += val;
          } else {
            outF[(size_t)m * N + n] = val;
          }
        }
      }
    }
}

// ---------------- differential flash attention ----------------
static __device__ __forceinline__ void online_update(v8f* s, float* mst, float* lst,
                                                     v8f* acc, unsigned short* pb,
                                                     int lh, int ll) {
#pragma unroll
  for (int i = 0; i < 8; ++i) {
    float rm = fmaxf(s[0][i], s[1][i]);
#pragma unroll
    for (int mk = 1; mk < 16; mk <<= 1) rm = fmaxf(rm, __shfl_xor(rm, mk, 32));
    const float nm = fmaxf(mst[i], rm);
    const float f = __expf(mst[i] - nm);
    const float p0 = __expf(s[0][i] - nm);
    const float p1 = __expf(s[1][i] - nm);
    float rs = p0 + p1;
#pragma unroll
    for (int mk = 1; mk < 16; mk <<= 1) rs += __shfl_xor(rs, mk, 32);
    lst[i] = lst[i] * f + rs;
    mst[i] = nm;
    acc[0][i] *= f; acc[1][i] *= f; acc[2][i] *= f; acc[3][i] *= f;
    const int pr = i + 8 * lh;
    pb[pr * 32 + ll] = f32_to_bf16(p0);
    pb[pr * 32 + 16 + ll] = f32_to_bf16(p1);
  }
}

// qkv: [token, 3C] bf16 (q at +0, k at +C, v at +2C)
__global__ __launch_bounds__(256)
void attn_kernel(const unsigned short* __restrict__ qkv,
                 const float* __restrict__ lam,
                 unsigned short* __restrict__ oB) {
  __shared__ __align__(16) unsigned short vt[64 * 40];       // [d][kv], pitch 40
  __shared__ __align__(16) unsigned short pbuf[8][2][16 * 32];
  const int t = threadIdx.x, lane = t & 31, w = t >> 5;
  const int b = blockIdx.x >> 4, h = blockIdx.x & 15;
  const int qb = blockIdx.y * 128;
  const int qw = qb + w * 16;                                // this wave's q rows
  const int lh = lane >> 4, ll = lane & 15;
  const float scale = 0.1767766952966369f;                   // 1/sqrt(32)
  const float lm = lam[h];

  Frag q1, q2;
  {
    const size_t rb = (size_t)(b * Tt + qw + ll) * C3 + h * HSs;
    q1.q[0] = *(const uint4*)(&qkv[rb + lh * 8]);
    q1.q[1] = *(const uint4*)(&qkv[rb + lh * 8 + 16]);
    q2.q[0] = *(const uint4*)(&qkv[rb + 32 + lh * 8]);
    q2.q[1] = *(const uint4*)(&qkv[rb + 32 + lh * 8 + 16]);
  }

  v8f acc1[4], acc2[4];
#pragma unroll
  for (int j = 0; j < 4; ++j) { acc1[j] = v8f{}; acc2[j] = v8f{}; }
  float m1[8], s1s[8], m2[8], s2s[8];
#pragma unroll
  for (int i = 0; i < 8; ++i) { m1[i] = -1e30f; m2[i] = -1e30f; s1s[i] = 0.f; s2s[i] = 0.f; }

  unsigned short* pb1 = &pbuf[w][0][0];
  unsigned short* pb2 = &pbuf[w][1][0];

  for (int kv0 = 0; kv0 < qb + 128; kv0 += 32) {
    {                                                        // stage V tile transposed
      const int kk = t >> 3, dc = (t & 7) * 8;
      uint4 d4 = *(const uint4*)(&qkv[(size_t)(b * Tt + kv0 + kk) * C3 + 2 * Cc + h * HSs + dc]);
      const unsigned short* sp = (const unsigned short*)&d4;
#pragma unroll
      for (int j = 0; j < 8; ++j) vt[(dc + j) * 40 + kk] = sp[j];
    }
    __syncthreads();

    v8f sc1[2], sc2[2];
#pragma unroll
    for (int ti = 0; ti < 2; ++ti) {
      const int n = kv0 + ti * 16 + ll;
      const size_t rb = (size_t)(b * Tt + n) * C3 + Cc + h * HSs;
      Frag kb1, kb2;
      kb1.q[0] = *(const uint4*)(&qkv[rb + lh * 8]);
      kb1.q[1] = *(const uint4*)(&qkv[rb + lh * 8 + 16]);
      kb2.q[0] = *(const uint4*)(&qkv[rb + 32 + lh * 8]);
      kb2.q[1] = *(const uint4*)(&qkv[rb + 32 + lh * 8 + 16]);
      v8f z = v8f{};
      sc1[ti] = wmma_bf16(q1.v, kb1.v, z);
      sc2[ti] = wmma_bf16(q2.v, kb2.v, z);
    }
    // scale + causal mask
#pragma unroll
    for (int ti = 0; ti < 2; ++ti)
#pragma unroll
      for (int i = 0; i < 8; ++i) {
        const int kv = kv0 + ti * 16 + ll;
        const int qrow = qw + i + 8 * lh;
        float a1 = sc1[ti][i] * scale, a2 = sc2[ti][i] * scale;
        if (kv > qrow) { a1 = -1e30f; a2 = -1e30f; }
        sc1[ti][i] = a1; sc2[ti][i] = a2;
      }

    online_update(sc1, m1, s1s, acc1, pb1, lh, ll);
    online_update(sc2, m2, s2s, acc2, pb2, lh, ll);

    Frag pa1, pa2;
    pa1.q[0] = *(const uint4*)(&pb1[ll * 32 + lh * 8]);
    pa1.q[1] = *(const uint4*)(&pb1[ll * 32 + lh * 8 + 16]);
    pa2.q[0] = *(const uint4*)(&pb2[ll * 32 + lh * 8]);
    pa2.q[1] = *(const uint4*)(&pb2[ll * 32 + lh * 8 + 16]);
#pragma unroll
    for (int j = 0; j < 4; ++j) {
      const int d = j * 16 + ll;
      Frag vb;
      vb.q[0] = *(const uint4*)(&vt[d * 40 + lh * 8]);
      vb.q[1] = *(const uint4*)(&vt[d * 40 + lh * 8 + 16]);
      acc1[j] = wmma_bf16(pa1.v, vb.v, acc1[j]);
      acc2[j] = wmma_bf16(pa2.v, vb.v, acc2[j]);
    }
    __syncthreads();
  }

#pragma unroll
  for (int j = 0; j < 4; ++j)
#pragma unroll
    for (int i = 0; i < 8; ++i) {
      const int row = qw + i + 8 * lh;
      const int d = j * 16 + ll;
      const float o = acc1[j][i] / s1s[i] - lm * acc2[j][i] / s2s[i];
      oB[(size_t)(b * Tt + row) * Cc + h * HSs + d] = f32_to_bf16(o);
    }
}

// ---------------- host orchestration ----------------
extern "C" void kernel_launch(void* const* d_in, const int* in_sizes, int n_in,
                              void* d_out, int out_size, void* d_ws, size_t ws_size,
                              hipStream_t stream) {
  const int*   idx   = (const int*)d_in[0];
  const float* wte   = (const float*)d_in[1];
  const float* wpe   = (const float*)d_in[2];
  const float* rms1  = (const float*)d_in[3];
  const float* rms2  = (const float*)d_in[4];
  const float* wq    = (const float*)d_in[5];
  const float* wk    = (const float*)d_in[6];
  const float* wv    = (const float*)d_in[7];
  const float* lq1   = (const float*)d_in[8];
  const float* lq2   = (const float*)d_in[9];
  const float* lk1   = (const float*)d_in[10];
  const float* lk2   = (const float*)d_in[11];
  const float* outw  = (const float*)d_in[12];
  const float* outb  = (const float*)d_in[13];
  const float* w1    = (const float*)d_in[14];
  const float* b1    = (const float*)d_in[15];
  const float* cw    = (const float*)d_in[16];
  const float* cb    = (const float*)d_in[17];
  const float* lnf   = (const float*)d_in[18];

  char* wsp = (char*)d_ws;
  auto alloc = [&](size_t bytes) -> char* {
    char* p = wsp;
    wsp += (bytes + 255) & ~(size_t)255;
    return p;
  };
  typedef unsigned short u16;
  u16* wteB  = (u16*)alloc((size_t)Vv * Cc * 2);
  u16* qkvT  = (u16*)alloc((size_t)Ll * C3 * Cc * 2);   // fused q/k/v weights, [3C, C] per layer
  u16* owT   = (u16*)alloc((size_t)Ll * Cc * Cc * 2);
  u16* w1T   = (u16*)alloc((size_t)Ll * Cc * C4 * 2);
  u16* cwT   = (u16*)alloc((size_t)Ll * C2 * Cc * 2);
  float* lamW = (float*)alloc((size_t)Ll * Hh * 4);
  float* x    = (float*)alloc((size_t)MT * Cc * 4);
  u16* hB     = (u16*)alloc((size_t)MT * Cc * 2);
  u16* qkvB   = (u16*)alloc((size_t)MT * C3 * 2);
  u16* oBuf   = (u16*)alloc((size_t)MT * Cc * 2);
  float* uBuf = (float*)alloc((size_t)MT * C4 * 4);
  u16* pBuf   = (u16*)alloc((size_t)MT * C2 * 2);
  u16* xfB    = (u16*)alloc((size_t)MT * Cc * 2);

  // weight precision conversion (amortized; bf16 working set then lives in 192MB L2)
  cvt_bf16_kernel<<<8192, 256, 0, stream>>>(wte, wteB, (long long)Vv * Cc);
  for (int l = 0; l < Ll; ++l) {
    transpose_cvt_kernel<<<dim3(Cc / 32, Cc / 32), 256, 0, stream>>>(
        wq + (size_t)l * Cc * Cc, qkvT + ((size_t)l * 3 + 0) * Cc * Cc, Cc, Cc);
    transpose_cvt_kernel<<<dim3(Cc / 32, Cc / 32), 256, 0, stream>>>(
        wk + (size_t)l * Cc * Cc, qkvT + ((size_t)l * 3 + 1) * Cc * Cc, Cc, Cc);
    transpose_cvt_kernel<<<dim3(Cc / 32, Cc / 32), 256, 0, stream>>>(
        wv + (size_t)l * Cc * Cc, qkvT + ((size_t)l * 3 + 2) * Cc * Cc, Cc, Cc);
    transpose_cvt_kernel<<<dim3(Cc / 32, Cc / 32), 256, 0, stream>>>(
        outw + (size_t)l * Cc * Cc, owT + (size_t)l * Cc * Cc, Cc, Cc);
    transpose_cvt_kernel<<<dim3(C4 / 32, Cc / 32), 256, 0, stream>>>(
        w1 + (size_t)l * Cc * C4, w1T + (size_t)l * Cc * C4, Cc, C4);
    transpose_cvt_kernel<<<dim3(Cc / 32, C2 / 32), 256, 0, stream>>>(
        cw + (size_t)l * C2 * Cc, cwT + (size_t)l * C2 * Cc, C2, Cc);
  }
  lam_kernel<<<1, 64, 0, stream>>>(lq1, lq2, lk1, lk2, lamW);
  embed_kernel<<<MT, 256, 0, stream>>>(idx, wte, wpe, x);

  const dim3 gNC(Cc / 128, MT / 128);
  const dim3 gN3(C3 / 128, MT / 128);
  const dim3 gN4(C4 / 128, MT / 128);
  for (int l = 0; l < Ll; ++l) {
    rmsnorm_kernel<<<MT, 256, 0, stream>>>(x, rms1 + (size_t)l * Cc, hB);
    gemm_kernel<false, false, true><<<gN3, 256, 0, stream>>>(
        hB, qkvT + (size_t)l * C3 * Cc, nullptr, nullptr, qkvB, MT, C3, Cc);
    attn_kernel<<<dim3(Bb * Hh, Tt / 128), 256, 0, stream>>>(
        qkvB, lamW + (size_t)l * Hh, oBuf);
    gemm_kernel<true, true, false><<<gNC, 256, 0, stream>>>(
        oBuf, owT + (size_t)l * Cc * Cc, outb + (size_t)l * Cc, x, nullptr, MT, Cc, Cc);
    rmsnorm_kernel<<<MT, 256, 0, stream>>>(x, rms2 + (size_t)l * Cc, hB);
    gemm_kernel<false, true, false><<<gN4, 256, 0, stream>>>(
        hB, w1T + (size_t)l * Cc * C4, b1 + (size_t)l * C4, uBuf, nullptr, MT, C4, Cc);
    swiglu_kernel<<<(int)(((size_t)MT * C2 + 255) / 256), 256, 0, stream>>>(uBuf, pBuf);
    gemm_kernel<true, true, false><<<gNC, 256, 0, stream>>>(
        pBuf, cwT + (size_t)l * C2 * Cc, cb + (size_t)l * Cc, x, nullptr, MT, Cc, C2);
  }

  final_ln_kernel<<<MT, 256, 0, stream>>>(x, lnf, xfB);
  gemm_kernel<false, false, false><<<dim3((Vv + 127) / 128, MT / 128), 256, 0, stream>>>(
      xfB, wteB, nullptr, (float*)d_out, nullptr, MT, Vv, Cc);
}
